// GroupedGAT_42889543418182
// MI455X (gfx1250) — matrix operand order repaired and checked by reading, not compiled, not run on hardware
//
#include <hip/hip_runtime.h>

typedef __bf16 bf16_t;
typedef bf16_t v16bf __attribute__((ext_vector_type(16)));
typedef float  v8f   __attribute__((ext_vector_type(8)));

union Frag {
  v16bf v;
  unsigned u[8];
  uint4 q[2];
};

__device__ __forceinline__ unsigned short f2bfbits(float f) {
  return __builtin_bit_cast(unsigned short, (bf16_t)f);   // hw cvt, RNE
}
__device__ __forceinline__ unsigned packbf2(float lo, float hi) {
  return ((unsigned)f2bfbits(hi) << 16) | (unsigned)f2bfbits(lo);
}
__device__ __forceinline__ float bf_lo(unsigned u) { return __builtin_bit_cast(float, u << 16); }
__device__ __forceinline__ float bf_hi(unsigned u) { return __builtin_bit_cast(float, u & 0xffff0000u); }

// ---------------------------------------------------------------------------
// f32 -> packed bf16 (one dword = two consecutive elements)
// ---------------------------------------------------------------------------
__global__ void f32_to_bf16(const float* __restrict__ src, unsigned* __restrict__ dst,
                            long totalPairs) {
  long i = (long)blockIdx.x * blockDim.x + threadIdx.x;
  if (i >= totalPairs) return;
  float2 f = ((const float2*)src)[i];
  dst[i] = packbf2(f.x, f.y);
}

// ---------------------------------------------------------------------------
// Pack f32 weights W[K][N] (nm matrices) into B fragments (this file's contiguous k order):
// [m][kt][nt][lane][v], lane%16 = N col, K = kt*32 + 16*(lane>>4) + 2*v.
// ---------------------------------------------------------------------------
__global__ void pack_b(const float* __restrict__ W, unsigned* __restrict__ dst,
                       int K, int N, int total) {
  int idx = blockIdx.x * blockDim.x + threadIdx.x;
  if (idx >= total) return;
  int lane = idx & 31;
  int r = idx >> 5;
  int NT = N >> 4, KT = K >> 5;
  int nt = r % NT; r /= NT;
  int kt = r % KT;
  int m  = r / KT;
  const float* Wm = W + (long)m * K * N;
  unsigned* d = dst + (long)idx * 8;
  int n  = nt * 16 + (lane & 15);
  int kh = (lane >> 4) << 4;
#pragma unroll
  for (int v = 0; v < 8; ++v) {
    int k = kt * 32 + kh + 2 * v;
    d[v] = packbf2(Wm[(long)k * N + n], Wm[(long)(k + 1) * N + n]);
  }
}

// ---------------------------------------------------------------------------
// WMMA GEMM: C = A x Bpk (+bias). A is bf16-packed [R][K/2] dwords, so an A
// fragment is two b128 loads per lane (no conversion in the hot loop).
// mode 0: identity rows. mode 1: rows 4m+{0,1}. mode 2: rows 4m+{2,3}.
// blockIdx.z = head. Output written as packed-bf16 elements (b16 stores).
// ---------------------------------------------------------------------------
template <int K, int N>
__global__ __launch_bounds__(128) void gemm_bf16(
    const unsigned* __restrict__ Abf, const unsigned* __restrict__ Bpk,
    const float* __restrict__ bias, unsigned short* __restrict__ Cb,
    int mode, int nTiles,
    long strideAh, long strideBh, long strideCh, long strideBiasH) {
  constexpr int KT = K / 32, NT = N / 16;
  int lane = threadIdx.x & 31;
  int wave = threadIdx.x >> 5;
  int rt = blockIdx.x * 4 + wave;
  if (rt >= nTiles) return;
  int head = blockIdx.z;
  Abf += (long)head * strideAh;
  Bpk += (long)head * strideBh;
  Cb  += (long)head * strideCh;
  int li = lane & 15, half = lane >> 4;
  auto maprow = [&](int i) -> long {
    if (mode == 0) return (long)rt * 16 + i;
    return ((long)rt * 8 + (i >> 1)) * 4 + (i & 1) + (mode == 2 ? 2 : 0);
  };
  const unsigned* ap = Abf + maprow(li) * (K / 2);

  v8f acc[NT];
#pragma unroll
  for (int nt = 0; nt < NT; ++nt) {
    v8f z = {0.f, 0.f, 0.f, 0.f, 0.f, 0.f, 0.f, 0.f};
    acc[nt] = z;
  }

  for (int kt = 0; kt < KT; ++kt) {
    Frag a;  // v0..7: k = 16*half + 0..15
    a.q[0] = *(const uint4*)(ap + kt * 16 + 8 * half);
    a.q[1] = *(const uint4*)(ap + kt * 16 + 8 * half + 4);
#pragma unroll
    for (int nt = 0; nt < NT; ++nt) {
      Frag b;
      const uint4* bp = (const uint4*)(Bpk + (((long)kt * NT + nt) * 32 + lane) * 8);
      b.q[0] = bp[0];
      b.q[1] = bp[1];
      acc[nt] = __builtin_amdgcn_wmma_f32_16x16x32_bf16(
          false, a.v, false, b.v, (short)0, acc[nt], false, false);
    }
  }

#pragma unroll
  for (int nt = 0; nt < NT; ++nt) {
    int col = nt * 16 + li;
    float bv = bias ? bias[(long)head * strideBiasH + col] : 0.f;
#pragma unroll
    for (int j = 0; j < 8; ++j) {
      long r = maprow(j + 8 * half);
      Cb[r * N + col] = f2bfbits(acc[nt][j] + bv);
    }
  }
}

// ---------------------------------------------------------------------------
// GAT attention over 4-node graphs, bf16 in / bf16 out (f32 math).
// 128 threads = 32 graphs per block; graph rows staged in LDS (stride 65).
// ---------------------------------------------------------------------------
__global__ __launch_bounds__(128) void gat_attn(
    const unsigned* __restrict__ Hbf, const float* __restrict__ a1g,
    const float* __restrict__ a2g, unsigned* __restrict__ OutBf,
    float* __restrict__ attn_out, int ngraphs,
    long strideHh, long strideOh, int aHeadStride) {  // dword strides
  __shared__ float s_h[128 * 65];
  __shared__ float s_a1[64], s_a2[64];
  int head = blockIdx.z;
  int t = threadIdx.x;
  if (t < 64)       s_a1[t]      = a1g[(long)head * aHeadStride + t];
  else if (t < 128) s_a2[t - 64] = a2g[(long)head * aHeadStride + (t - 64)];

  int node = t & 3;
  int g = blockIdx.x * 32 + (t >> 2);
  bool active = g < ngraphs;

  const unsigned* hp = Hbf + (long)head * strideHh + ((long)g * 4 + node) * 32;
  float* myrow = s_h + t * 65;
  if (active) {
#pragma unroll
    for (int d = 0; d < 32; ++d) {
      unsigned u = hp[d];
      myrow[2 * d]     = bf_lo(u);
      myrow[2 * d + 1] = bf_hi(u);
    }
  }
  __syncthreads();

  float A1 = 0.f, A2 = 0.f;
#pragma unroll 8
  for (int k = 0; k < 64; ++k) {
    float v = myrow[k];
    A1 += v * s_a1[k];
    A2 += v * s_a2[k];
  }
  int base = (t & 31) & ~3;
  float e[4], w[4], m = -1e30f;
#pragma unroll
  for (int j = 0; j < 4; ++j) {
    float A2j = __shfl(A2, base + j, 32);
    float x = A1 + A2j;
    e[j] = x > 0.f ? x : 0.2f * x;          // leaky relu, alpha=0.2
    m = fmaxf(m, e[j]);
  }
  float s = 0.f;
#pragma unroll
  for (int j = 0; j < 4; ++j) { w[j] = __expf(e[j] - m); s += w[j]; }
  float inv = 1.f / s;
#pragma unroll
  for (int j = 0; j < 4; ++j) w[j] *= inv;

  if (active && attn_out) {
    float* ar = attn_out + ((long)g * 4 + node) * 4;
#pragma unroll
    for (int j = 0; j < 4; ++j) ar[j] = w[j];
  }
  if (active) {
    const float* g0 = s_h + (t & ~3) * 65;
    unsigned* orow = OutBf + (long)head * strideOh + ((long)g * 4 + node) * 32;
#pragma unroll 4
    for (int d = 0; d < 32; ++d) {
      float v0 = w[0] * g0[2 * d]     + w[1] * g0[65 + 2 * d]     + w[2] * g0[130 + 2 * d]     + w[3] * g0[195 + 2 * d];
      float v1 = w[0] * g0[2 * d + 1] + w[1] * g0[65 + 2 * d + 1] + w[2] * g0[130 + 2 * d + 1] + w[3] * g0[195 + 2 * d + 1];
      v0 = v0 > 0.f ? v0 : (__expf(v0) - 1.f);   // elu
      v1 = v1 > 0.f ? v1 : (__expf(v1) - 1.f);
      orow[d] = packbf2(v0, v1);
    }
  }
}

// X2[b][t][h*64+c] = XP[h][bg][p][c] + XG[h][bg][c]   (all bf16 packed)
__global__ void add_concat(const unsigned* __restrict__ XP, const unsigned* __restrict__ XG,
                           unsigned* __restrict__ X2, long totalPairs) {
  long idx = (long)blockIdx.x * blockDim.x + threadIdx.x;
  if (idx >= totalPairs) return;
  int cp = (int)(idx & 31);
  long tmp = idx >> 5;
  int tt = (int)(tmp & 15); tmp >>= 4;
  long b = tmp % 16000;
  long h = tmp / 16000;
  int gg = tt >> 2, p = tt & 3;
  long bg = b * 4 + gg;
  unsigned up = XP[h * 8192000L + (bg * 4 + p) * 32 + cp];
  unsigned ug = XG[h * 2048000L + bg * 32 + cp];
  X2[(b * 16 + tt) * 128 + h * 32 + cp] = packbf2(bf_lo(up) + bf_lo(ug), bf_hi(up) + bf_hi(ug));
}

// out[b][t][c] = XPo + broadcast(XGo)  (bf16 in, f32 out)
__global__ void add_final(const unsigned* __restrict__ XP, const unsigned* __restrict__ XG,
                          float* __restrict__ O, long totalPairs) {
  long idx = (long)blockIdx.x * blockDim.x + threadIdx.x;
  if (idx >= totalPairs) return;
  int cp = (int)(idx & 31);
  long tmp = idx >> 5;
  int tt = (int)(tmp & 15);
  long b = tmp >> 4;
  long bg = b * 4 + (tt >> 2);
  int p = tt & 3;
  unsigned up = XP[(bg * 4 + p) * 32 + cp];
  unsigned ug = XG[bg * 32 + cp];
  float2 o;
  o.x = bf_lo(up) + bf_lo(ug);
  o.y = bf_hi(up) + bf_hi(ug);
  ((float2*)O)[idx] = o;
}

extern "C" void kernel_launch(void* const* d_in, const int* in_sizes, int n_in,
                              void* d_out, int out_size, void* d_ws, size_t ws_size,
                              hipStream_t stream) {
  (void)in_sizes; (void)n_in; (void)out_size; (void)ws_size;
  const float* inp = (const float*)d_in[0];
  // d_in[1] = taxi_size == 2 (hardcoded)
  const float* W1  = (const float*)d_in[2];
  const float* W2  = (const float*)d_in[3];
  const float* a1  = (const float*)d_in[4];
  const float* a2  = (const float*)d_in[5];
  const float* gW  = (const float*)d_in[6];
  const float* gb  = (const float*)d_in[7];
  const float* oW1 = (const float*)d_in[8];
  const float* oW2 = (const float*)d_in[9];
  const float* oa1 = (const float*)d_in[10];
  const float* oa2 = (const float*)d_in[11];
  const float* ogW = (const float*)d_in[12];
  const float* ogb = (const float*)d_in[13];

  char* ws = (char*)d_ws;
  size_t off = 0;
  auto alloc = [&](size_t bytes) { size_t o = off; off += (bytes + 255) & ~(size_t)255; return o; };

  // packed bf16 weight fragments
  unsigned* pkW1  = (unsigned*)(ws + alloc(4 * 2048 * 4));
  unsigned* pkW2  = (unsigned*)(ws + alloc(4 * 2048 * 4));
  unsigned* pkGW  = (unsigned*)(ws + alloc(4 * 8192 * 4));
  unsigned* pkOW1 = (unsigned*)(ws + alloc(8192 * 4));
  unsigned* pkOW2 = (unsigned*)(ws + alloc(8192 * 4));
  unsigned* pkOGW = (unsigned*)(ws + alloc(32768 * 4));

  // bf16 activation buffers (dword = 2 elements), reused along the chain
  unsigned* inpBf = (unsigned*)(ws + alloc(32768000));   // [256000][32]
  unsigned* bufA  = (unsigned*)(ws + alloc(131072000));  // H1 [4][256000][32] -> X2 [256000][128]
  unsigned* bufB  = (unsigned*)(ws + alloc(131072000));  // XP1 [4][64000][128]
  unsigned* bufC  = (unsigned*)(ws + alloc(32768000));   // XGin1 -> H1o -> XGin_o
  unsigned* bufD  = (unsigned*)(ws + alloc(32768000));   // H2 -> XPo [64000][128]
  unsigned* bufE  = (unsigned*)(ws + alloc(32768000));   // XG1 -> H2o | XGo
  unsigned* H2o = bufE;
  unsigned* XGo = (unsigned*)((char*)bufE + 16384000);

  float* outMain = (float*)d_out;        // [16000][16][64]
  float* outAttn = outMain + 16384000;   // [64000][4][4]
  float* outGatt = outMain + 17408000;   // [16000][4][4]

  auto packL = [&](const float* W, unsigned* dst, int K, int N, int nm) {
    int total = nm * (K / 32) * (N / 16) * 32;
    pack_b<<<(total + 255) / 256, 256, 0, stream>>>(W, dst, K, N, total);
  };
  packL(W1, pkW1, 64, 64, 4);
  packL(W2, pkW2, 64, 64, 4);
  packL(gW, pkGW, 256, 64, 4);
  packL(oW1, pkOW1, 256, 64, 1);
  packL(oW2, pkOW2, 256, 64, 1);
  packL(ogW, pkOGW, 256, 256, 1);

  // input -> bf16
  f32_to_bf16<<<32000, 256, 0, stream>>>(inp, inpBf, 8192000L);

  // --- Stage A -------------------------------------------------------------
  // G1: inpBf @ W1/W2 -> H1 (bufA), per head
  {
    dim3 grid(2000, 1, 4);  // 8000 tiles
    gemm_bf16<64, 64><<<grid, 128, 0, stream>>>(inpBf, pkW1, nullptr, (unsigned short*)bufA, 1, 8000, 0L, 2048L, 16384000L, 0L);
    gemm_bf16<64, 64><<<grid, 128, 0, stream>>>(inpBf, pkW2, nullptr, (unsigned short*)bufA, 2, 8000, 0L, 2048L, 16384000L, 0L);
  }
  // Att1: H1 -> XP1 (bufB)
  gat_attn<<<dim3(2000, 1, 4), 128, 0, stream>>>(bufA, a1, a2, bufB, nullptr, 64000, 8192000L, 8192000L, 64);
  // G2: XP1 @ gW + gb -> XGin1 (bufC)
  gemm_bf16<256, 64><<<dim3(1000, 1, 4), 128, 0, stream>>>(bufB, pkGW, gb, (unsigned short*)bufC, 0, 4000, 8192000L, 8192L, 4096000L, 64L);
  // G3: XGin1 @ W1/W2 -> H2 (bufD)
  {
    dim3 grid(500, 1, 4);  // 2000 tiles
    gemm_bf16<64, 64><<<grid, 128, 0, stream>>>(bufC, pkW1, nullptr, (unsigned short*)bufD, 1, 2000, 2048000L, 2048L, 4096000L, 0L);
    gemm_bf16<64, 64><<<grid, 128, 0, stream>>>(bufC, pkW2, nullptr, (unsigned short*)bufD, 2, 2000, 2048000L, 2048L, 4096000L, 0L);
  }
  // Att2: H2 -> XG1 (bufE)
  gat_attn<<<dim3(500, 1, 4), 128, 0, stream>>>(bufD, a1, a2, bufE, nullptr, 16000, 2048000L, 2048000L, 64);
  // concat-add heads into X2 (bufA)
  add_concat<<<128000, 256, 0, stream>>>(bufB, bufE, bufA, 32768000L);

  // --- Stage B -------------------------------------------------------------
  // G4: X2 @ oW1/oW2 -> H1o (bufC)
  {
    dim3 grid(2000, 1, 1);  // 8000 tiles
    gemm_bf16<256, 64><<<grid, 128, 0, stream>>>(bufA, pkOW1, nullptr, (unsigned short*)bufC, 1, 8000, 0L, 0L, 0L, 0L);
    gemm_bf16<256, 64><<<grid, 128, 0, stream>>>(bufA, pkOW2, nullptr, (unsigned short*)bufC, 2, 8000, 0L, 0L, 0L, 0L);
  }
  // Att3: H1o -> XPo (bufD), write attn output
  gat_attn<<<dim3(2000, 1, 1), 128, 0, stream>>>(bufC, oa1, oa2, bufD, outAttn, 64000, 0L, 0L, 0);
  // G5: XPo @ ogW + ogb -> XGin_o (bufC)
  gemm_bf16<256, 256><<<dim3(1000, 1, 1), 128, 0, stream>>>(bufD, pkOGW, ogb, (unsigned short*)bufC, 0, 4000, 0L, 0L, 0L, 0L);
  // G6: XGin_o @ oW1/oW2 -> H2o (bufE)
  {
    dim3 grid(500, 1, 1);  // 2000 tiles
    gemm_bf16<256, 64><<<grid, 128, 0, stream>>>(bufC, pkOW1, nullptr, (unsigned short*)H2o, 1, 2000, 0L, 0L, 0L, 0L);
    gemm_bf16<256, 64><<<grid, 128, 0, stream>>>(bufC, pkOW2, nullptr, (unsigned short*)H2o, 2, 2000, 0L, 0L, 0L, 0L);
  }
  // Att4: H2o -> XGo, write gattn output
  gat_attn<<<dim3(500, 1, 1), 128, 0, stream>>>(H2o, oa1, oa2, XGo, outGatt, 16000, 0L, 0L, 0);
  // final: out = XPo + broadcast(XGo)
  add_final<<<32000, 256, 0, stream>>>(bufD, XGo, outMain, 8192000L);
}
